// SparseAttention_31825707663464
// MI455X (gfx1250) — matrix operand order repaired and checked by reading, not compiled, not run on hardware
//
#include <hip/hip_runtime.h>

#define BATCH 4
#define HEADS 16
#define SEQ   4096
#define DIM   64
#define NNZ   512
#define KC    32            // keys per chunk (one TDM-gathered tile pair, double buffered)
#define NC    (NNZ / KC)    // 16 chunks
#define QTILE 128           // query rows per block (8 waves * 16)
#define WAVES 8

typedef __attribute__((ext_vector_type(16))) __bf16   v16bf;
typedef __attribute__((ext_vector_type(8)))  __bf16   v8bf;
typedef __attribute__((ext_vector_type(8)))  float    v8f;
typedef __attribute__((ext_vector_type(4)))  unsigned u32x4;
typedef __attribute__((ext_vector_type(8)))  unsigned u32x8;

// native fptrunc: gfx1250 bf16 VALU -> v_cvt_pk_bf16_f32
__device__ __forceinline__ __bf16 f2bf(float x) { return (__bf16)x; }

// single-instruction max (avoids fmaxf's canonicalize self-max chains)
__device__ __forceinline__ float fmaxq(float a, float b) {
  float d;
  asm("v_max_num_f32 %0, %1, %2" : "=v"(d) : "v"(a), "v"(b));
  return d;
}

// 16-lane-group xor butterfly step via raw v_permlane16_b32 (no tied-old copy)
template <unsigned S0, unsigned S1>
__device__ __forceinline__ float lanex(float x) {
  float d;
  asm("v_permlane16_b32 %0, %1, %2, %3" : "=v"(d) : "v"(x), "s"(S0), "s"(S1));
  return d;
}
#define XOR1(x) lanex<0x67452301u, 0xEFCDAB89u>(x)
#define XOR2(x) lanex<0x54761032u, 0xDCFE98BAu>(x)
#define XOR4(x) lanex<0x32107654u, 0xBA98FEDCu>(x)
#define XOR8(x) lanex<0xFEDCBA98u, 0x76543210u>(x)

__device__ __forceinline__ float rmax16(float v) {
  v = fmaxq(v, XOR1(v)); v = fmaxq(v, XOR2(v));
  v = fmaxq(v, XOR4(v)); v = fmaxq(v, XOR8(v));
  return v;
}
__device__ __forceinline__ float rsum16(float v) {
  v += XOR1(v); v += XOR2(v); v += XOR4(v); v += XOR8(v);
  return v;
}

__device__ __forceinline__ float fexp2(float x) {
#if __has_builtin(__builtin_amdgcn_exp2f)
  return __builtin_amdgcn_exp2f(x);      // v_exp_f32
#else
  return exp2f(x);
#endif
}
__device__ __forceinline__ float frcp(float x) {
#if __has_builtin(__builtin_amdgcn_rcpf)
  return __builtin_amdgcn_rcpf(x);       // v_rcp_f32
#else
  return 1.f / x;
#endif
}

// TDM gather-mode load: 16 indexed rows of 64 f32 each -> LDS.
// D# per CDNA5 ISA 8.3-8.7: group0 = {flags, lds_addr, global_addr, type=2},
// group1 = {data_size=4B, tensor/tile dims, dim0 stride}, groups 2/3 = 16x 16-bit row indices.
__device__ __forceinline__ void tdm_gather16(unsigned lds_off,
                                             unsigned long long gaddr,
                                             const int* idx) {
  u32x4 g0, g2, g3;
  u32x8 g1;
  g0[0] = 0x80000001u;                                   // gather_mode=1, 16-bit idx, count=1
  g0[1] = lds_off;                                       // LDS byte address
  g0[2] = (unsigned)gaddr;                               // global_addr[31:0]
  g0[3] = (unsigned)((gaddr >> 32) & 0x01FFFFFFull)      // global_addr[56:32]
        | 0x80000000u;                                   // type=2 ("image")
  g1[0] = 0x00020000u;                                   // data_size = 4 bytes
  g1[1] = (unsigned)DIM << 16;                           // tensor_dim0 = 64 (bits 79:48 lo)
  g1[2] = (unsigned)SEQ << 16;                           // tensor_dim1 = 4096 (bits 111:80 lo)
  g1[3] = (unsigned)DIM << 16;                           // tile_dim0 = 64 (bits 127:112)
  g1[4] = 16u;                                           // tile_dim1 = #gather indices
  g1[5] = (unsigned)DIM;                                 // tensor_dim0_stride = 64
  g1[6] = 0u;
  g1[7] = 0u;
  #pragma unroll
  for (int j = 0; j < 4; ++j) {                          // pack sorted indices, 2 per dword;
    unsigned p2 = ((unsigned)idx[2 * j]     & 0xFFFFu) | ((unsigned)idx[2 * j + 1]     << 16);
    unsigned p3 = ((unsigned)idx[8 + 2 * j] & 0xFFFFu) | ((unsigned)idx[8 + 2 * j + 1] << 16);
    // indices are workgroup-uniform; force SGPR register class for the "s" asm constraint
    g2[j] = (unsigned)__builtin_amdgcn_readfirstlane((int)p2);
    g3[j] = (unsigned)__builtin_amdgcn_readfirstlane((int)p3);
  }
  asm volatile("tensor_load_to_lds %0, %1, %2, %3"
               :: "s"(g0), "s"(g1), "s"(g2), "s"(g3)
               : "memory");
}

__global__ __launch_bounds__(256)
void sparse_attn_wmma(const float* __restrict__ q,
                      const float* __restrict__ kk,
                      const float* __restrict__ vv,
                      const int*   __restrict__ mask,
                      float* __restrict__ out)
{
  __shared__ __align__(32) float  Kstage[2][KC * DIM];        // TDM f32 staging, 2x8KB
  __shared__ __align__(32) float  Vstage[2][KC * DIM];        // TDM f32 staging, 2x8KB
  __shared__ __align__(32) __bf16 Kbf[KC * DIM];              // bf16 [key][d], 4KB
  __shared__ __align__(32) __bf16 VbfT[DIM * KC];             // bf16 [d][key] (transposed), 4KB
  __shared__ __align__(32) __bf16 Psh[WAVES * 16 * KC];       // per-wave P, 8KB

  const int tid  = threadIdx.x;
  const int wave = tid >> 5;
  const int lane = tid & 31;
  const int ln   = lane & 15;
  const bool hiL = lane >= 16;

  const int bh = blockIdx.y;                 // b*H + h
  const int h  = bh & (HEADS - 1);
  const int q0 = blockIdx.x * QTILE + wave * 16;

  const float* qbase = q  + (size_t)bh * SEQ * DIM;
  const float* kbase = kk + (size_t)bh * SEQ * DIM;
  const float* vbase = vv + (size_t)bh * SEQ * DIM;
  const int*   mbase = mask + h * NNZ;

  // LDS byte offsets (generic LDS address: low 32 bits are the LDS offset)
  const unsigned kshOff = (unsigned)(size_t)(void*)&Kstage[0][0];
  const unsigned vshOff = (unsigned)(size_t)(void*)&Vstage[0][0];
  const unsigned long long kga = (unsigned long long)(size_t)(const void*)kbase;
  const unsigned long long vga = (unsigned long long)(size_t)(const void*)vbase;

  // ---- Q fragments: A-layout 16x32 bf16, two fragments covering d=0..63.
  // Softmax scale and log2(e) folded into Q before bf16 conversion (base-2 softmax).
  const float qscale = 0.125f * 1.44269504088896340736f;
  v16bf qfrag[2];
  {
    const float* qr = qbase + (size_t)(q0 + ln) * DIM;
    #pragma unroll
    for (int f = 0; f < 2; ++f) {
      const int k0 = f * 32 + (hiL ? 8 : 0);
      const float4 a0 = *reinterpret_cast<const float4*>(qr + k0);
      const float4 a1 = *reinterpret_cast<const float4*>(qr + k0 + 4);
      const float4 b0 = *reinterpret_cast<const float4*>(qr + k0 + 16);
      const float4 b1 = *reinterpret_cast<const float4*>(qr + k0 + 20);
      qfrag[f][0]=f2bf(a0.x*qscale);  qfrag[f][1]=f2bf(a0.y*qscale);
      qfrag[f][2]=f2bf(a0.z*qscale);  qfrag[f][3]=f2bf(a0.w*qscale);
      qfrag[f][4]=f2bf(a1.x*qscale);  qfrag[f][5]=f2bf(a1.y*qscale);
      qfrag[f][6]=f2bf(a1.z*qscale);  qfrag[f][7]=f2bf(a1.w*qscale);
      qfrag[f][8]=f2bf(b0.x*qscale);  qfrag[f][9]=f2bf(b0.y*qscale);
      qfrag[f][10]=f2bf(b0.z*qscale); qfrag[f][11]=f2bf(b0.w*qscale);
      qfrag[f][12]=f2bf(b1.x*qscale); qfrag[f][13]=f2bf(b1.y*qscale);
      qfrag[f][14]=f2bf(b1.z*qscale); qfrag[f][15]=f2bf(b1.w*qscale);
    }
  }

  v8f acc[4];
  #pragma unroll
  for (int t = 0; t < 4; ++t)
    #pragma unroll
    for (int r = 0; r < 8; ++r) acc[t][r] = 0.f;

  float m_run[8], l_run[8];   // running max / sum in log2 domain
  #pragma unroll
  for (int r = 0; r < 8; ++r) { m_run[r] = -1e30f; l_run[r] = 0.f; }

  // ---- prologue: async-gather chunk 0 into staging buffer 0 (wave 0 only)
  if (wave == 0) {
    #pragma unroll
    for (int dblk = 0; dblk < 2; ++dblk) {
      const int* mr = mbase + dblk * 16;
      tdm_gather16(kshOff + (unsigned)(dblk * 16 * DIM * 4), kga, mr);
      tdm_gather16(vshOff + (unsigned)(dblk * 16 * DIM * 4), vga, mr);
    }
  }

  for (int c = 0; c < NC; ++c) {
    const int buf = c & 1;
    if (wave == 0) __builtin_amdgcn_s_wait_tensorcnt(0);   // chunk c landed in LDS
    __syncthreads();                                        // staged data visible; bf16 bufs free
    if (wave == 0 && (c + 1) < NC) {
      const unsigned bo = (unsigned)((buf ^ 1) * KC * DIM * 4);
      #pragma unroll
      for (int dblk = 0; dblk < 2; ++dblk) {
        const int* mr = mbase + (c + 1) * KC + dblk * 16;
        tdm_gather16(kshOff + bo + (unsigned)(dblk * 16 * DIM * 4), kga, mr);  // overlaps compute
        tdm_gather16(vshOff + bo + (unsigned)(dblk * 16 * DIM * 4), vga, mr);
      }
    }

    // ---- cooperative f32 -> bf16 conversion (each element converted exactly once)
    {
      const float* Kc = &Kstage[buf][0];
      const float* Vc = &Vstage[buf][0];
      const int e = tid * 8;                 // 256 threads x 8 elements = 2048 (= KC*DIM)
      // K: row-major [key][d]
      const float4 ka = *reinterpret_cast<const float4*>(Kc + e);
      const float4 kb = *reinterpret_cast<const float4*>(Kc + e + 4);
      v8bf kvv;
      kvv[0]=f2bf(ka.x); kvv[1]=f2bf(ka.y); kvv[2]=f2bf(ka.z); kvv[3]=f2bf(ka.w);
      kvv[4]=f2bf(kb.x); kvv[5]=f2bf(kb.y); kvv[6]=f2bf(kb.z); kvv[7]=f2bf(kb.w);
      *reinterpret_cast<v8bf*>(&Kbf[e]) = kvv;
      // V: transposed [d][key] so gemm-2 B fragments are contiguous along the key contraction
      const int key = e >> 6;
      const int d0  = e & 63;
      const float4 va = *reinterpret_cast<const float4*>(Vc + e);
      const float4 vb = *reinterpret_cast<const float4*>(Vc + e + 4);
      VbfT[(d0 + 0) * KC + key] = f2bf(va.x);
      VbfT[(d0 + 1) * KC + key] = f2bf(va.y);
      VbfT[(d0 + 2) * KC + key] = f2bf(va.z);
      VbfT[(d0 + 3) * KC + key] = f2bf(va.w);
      VbfT[(d0 + 4) * KC + key] = f2bf(vb.x);
      VbfT[(d0 + 5) * KC + key] = f2bf(vb.y);
      VbfT[(d0 + 6) * KC + key] = f2bf(vb.z);
      VbfT[(d0 + 7) * KC + key] = f2bf(vb.w);
    }
    __syncthreads();                         // bf16 buffers ready for all waves

    // ---- S = Q * K^T (log2-domain logits) for 16 rows x 32 keys
    float s[2][8];
    #pragma unroll
    for (int nt = 0; nt < 2; ++nt) {
      v8f cacc;
      #pragma unroll
      for (int r = 0; r < 8; ++r) cacc[r] = 0.f;
      #pragma unroll
      for (int f = 0; f < 2; ++f) {
        // B-layout: lane = key column; halves = contraction d (table order: 8h.., 16+8h..)
        const __bf16* kr = &Kbf[(nt * 16 + ln) * DIM + f * 32 + (hiL ? 8 : 0)];
        const v8bf klo = *reinterpret_cast<const v8bf*>(kr);
        const v8bf khi = *reinterpret_cast<const v8bf*>(kr + 16);
        v16bf bfrag;
        #pragma unroll
        for (int j = 0; j < 8; ++j) { bfrag[j] = klo[j]; bfrag[j + 8] = khi[j]; }
        cacc = __builtin_amdgcn_wmma_f32_16x16x32_bf16(false, qfrag[f], false, bfrag,
                                                       (short)0, cacc, false, false);
      }
      #pragma unroll
      for (int r = 0; r < 8; ++r) s[nt][r] = cacc[r];
    }

    // ---- online softmax, base-2 (row stats within each 16-lane group)
    #pragma unroll
    for (int r = 0; r < 8; ++r) {
      const float cm = rmax16(fmaxq(s[0][r], s[1][r]));
      const float mn    = fmaxq(m_run[r], cm);
      const float alpha = fexp2(m_run[r] - mn);
      m_run[r] = mn;
      const float p0 = fexp2(s[0][r] - mn);
      const float p1 = fexp2(s[1][r] - mn);
      s[0][r] = p0; s[1][r] = p1;
      const float rs = rsum16(p0 + p1);
      l_run[r] = l_run[r] * alpha + rs;
      #pragma unroll
      for (int t = 0; t < 4; ++t) acc[t][r] *= alpha;
    }

    // ---- P (C-layout f32) -> wave-private LDS -> single A-layout bf16 fragment (16x32)
    {
      __bf16* pw = &Psh[wave * 16 * KC];
      #pragma unroll
      for (int nt = 0; nt < 2; ++nt)
        #pragma unroll
        for (int r = 0; r < 8; ++r)
          pw[(r + (hiL ? 8 : 0)) * KC + nt * 16 + ln] = f2bf(s[nt][r]);
    }
    __builtin_amdgcn_wave_barrier();   // LDS in-order per wave; just block reordering

    v16bf pfrag;
    {
      const __bf16* pr = &Psh[wave * 16 * KC + ln * KC];
      const int k0 = hiL ? 16 : 0;
      const v8bf lo = *reinterpret_cast<const v8bf*>(pr + k0);
      const v8bf hv = *reinterpret_cast<const v8bf*>(pr + k0 + 8);
      #pragma unroll
      for (int j = 0; j < 8; ++j) { pfrag[j] = lo[j]; pfrag[j + 8] = hv[j]; }
    }

    // ---- O += P * V  (contraction = 32 keys -> 1 WMMA per d-tile; B from VbfT, contiguous)
    #pragma unroll
    for (int dt = 0; dt < 4; ++dt) {
      const v16bf bfrag = *reinterpret_cast<const v16bf*>(
          &VbfT[(dt * 16 + ln) * KC + (hiL ? 16 : 0)]);
      acc[dt] = __builtin_amdgcn_wmma_f32_16x16x32_bf16(false, pfrag, false, bfrag,
                                                        (short)0, acc[dt], false, false);
    }
  }

  // ---- normalize and store (C-layout: VGPR r -> row r / r+8, lane -> d)
  float* ob = out + ((size_t)bh * SEQ + q0) * DIM;
  #pragma unroll
  for (int r = 0; r < 8; ++r) {
    const float inv = frcp(l_run[r]);
    const int row = r + (hiL ? 8 : 0);
    #pragma unroll
    for (int dt = 0; dt < 4; ++dt)
      ob[(size_t)row * DIM + dt * 16 + ln] = acc[dt][r] * inv;
  }
}

extern "C" void kernel_launch(void* const* d_in, const int* in_sizes, int n_in,
                              void* d_out, int out_size, void* d_ws, size_t ws_size,
                              hipStream_t stream) {
  (void)in_sizes; (void)n_in; (void)out_size; (void)d_ws; (void)ws_size;
  const float* q = (const float*)d_in[0];
  const float* k = (const float*)d_in[1];
  const float* v = (const float*)d_in[2];
  const int*   m = (const int*)d_in[3];
  dim3 grid(SEQ / QTILE, BATCH * HEADS);
  sparse_attn_wmma<<<grid, dim3(256), 0, stream>>>(q, k, v, m, (float*)d_out);
}
